// HeteroSAGEAttention_59854664237758
// MI455X (gfx1250) — matrix operand order repaired and checked by reading, not compiled, not run on hardware
//
#include <hip/hip_runtime.h>
#include <hip/hip_bf16.h>

typedef __attribute__((ext_vector_type(16))) __bf16 v16bf;
typedef __attribute__((ext_vector_type(8)))  float  v8f;

#define NU_  100000
#define NS_  100000
#define EDG_ 500000
#define HCH_ 128

// ---------------------------------------------------------------------------
// GEMM  C[M,N] = A[M,K] * W[K,N] (+bias), fp32 in/out, bf16 WMMA core.
// Block = 256 threads = 8 waves; each wave owns a 16-row strip and all N/16
// column tiles. W is staged to LDS in bf16, column-major (wlds[n*K+k]) so a
// lane's 16-element B fragment is one contiguous 32-byte LDS read.
// ---------------------------------------------------------------------------
template <int K, int N>
__global__ __launch_bounds__(256) void gemm_bf16_wmma(
    const float* __restrict__ A, const float* __restrict__ W,
    const float* __restrict__ bias, float* __restrict__ C, int M)
{
  __shared__ __bf16 wlds[128 * 128];      // 32 KB, col-major [n][k]
  const int tid = threadIdx.x;
  for (int i = tid; i < K * N; i += 256) {
    int k = i / N, n = i % N;
    wlds[n * K + k] = (__bf16)W[i];
  }
  __syncthreads();

  const int wave = tid >> 5, lane = tid & 31;
  const int half = lane >> 4, l16 = lane & 15;
  const int mbase = blockIdx.x * 128 + wave * 16;
  const int row   = mbase + l16;          // A row this lane loads
  const bool rowok = row < M;
  constexpr int NT = N / 16;

  v8f zero = {};
  v8f acc[NT];
#pragma unroll
  for (int nt = 0; nt < NT; ++nt) acc[nt] = zero;

#pragma unroll
  for (int kt = 0; kt < K; kt += 32) {
    // A fragment (contiguous k, matching B): lanes 0-15 hold K=kt+0..15,
    // lanes 16-31 hold K=kt+16..31 for row M = l16.
    float4 a0 = make_float4(0.f, 0.f, 0.f, 0.f);
    float4 a1 = a0, a2 = a0, a3 = a0;
    if (rowok) {
      const float4* p = (const float4*)(A + (size_t)row * K);
      const int k0 = (kt + half * 16) >> 2;
      const int k1 = (kt + half * 16 + 8) >> 2;
      a0 = p[k0]; a1 = p[k0 + 1];
      a2 = p[k1]; a3 = p[k1 + 1];
    }
    v16bf af;
    af[0]  = (__bf16)a0.x; af[1]  = (__bf16)a0.y; af[2]  = (__bf16)a0.z; af[3]  = (__bf16)a0.w;
    af[4]  = (__bf16)a1.x; af[5]  = (__bf16)a1.y; af[6]  = (__bf16)a1.z; af[7]  = (__bf16)a1.w;
    af[8]  = (__bf16)a2.x; af[9]  = (__bf16)a2.y; af[10] = (__bf16)a2.z; af[11] = (__bf16)a2.w;
    af[12] = (__bf16)a3.x; af[13] = (__bf16)a3.y; af[14] = (__bf16)a3.z; af[15] = (__bf16)a3.w;

#pragma unroll
    for (int nt = 0; nt < NT; ++nt) {
      // B fragment: lane column n = nt*16+l16, K run = kt + half*16 .. +15
      const v16bf bf =
          *(const v16bf*)&wlds[(size_t)(nt * 16 + l16) * K + kt + half * 16];
      acc[nt] = __builtin_amdgcn_wmma_f32_16x16x32_bf16(
          false, af, false, bf, (short)0, acc[nt], false, false);
    }
  }

  // C/D layout: VGPR r, lane l -> M = mbase + (l>=16?8:0) + r, Ncol = l%16
#pragma unroll
  for (int nt = 0; nt < NT; ++nt) {
#pragma unroll
    for (int r = 0; r < 8; ++r) {
      const int m = mbase + half * 8 + r;
      if (m < M) {
        const int n = nt * 16 + l16;
        float v = acc[nt][r];
        if (bias) v += bias[n];
        C[(size_t)m * N + n] = v;
      }
    }
  }
}

// ---------------------------------------------------------------------------
// Per-node attention dot: p[n] = dot(h[n, 0:128], a[0:128]).  Wave per node.
// ---------------------------------------------------------------------------
__global__ __launch_bounds__(256) void nodedot(
    const float* __restrict__ h, const float* __restrict__ a,
    float* __restrict__ p, int N)
{
  const int node = blockIdx.x * 8 + (threadIdx.x >> 5);
  if (node >= N) return;
  const int lane = threadIdx.x & 31;
  const float4 hv = ((const float4*)(h + (size_t)node * HCH_))[lane];
  const float4 av = ((const float4*)a)[lane];
  float s = hv.x * av.x + hv.y * av.y + hv.z * av.z + hv.w * av.w;
#pragma unroll
  for (int off = 16; off > 0; off >>= 1) s += __shfl_down(s, off, 32);
  if (lane == 0) p[node] = s;
}

// ---------------------------------------------------------------------------
// Edge scatter (segment_sum with atomics). Wave per edge.
// Reference quirk preserved: g2 row = esrc[esrc[e]] (double gather).
// ---------------------------------------------------------------------------
__global__ __launch_bounds__(256) void edge_scatter(
    const int* __restrict__ esrc, const int* __restrict__ etgt,
    const float* __restrict__ src_h, const float* __restrict__ p_src,
    const float* __restrict__ p_tgt, float* __restrict__ num,
    float* __restrict__ den, int E)
{
  const int e = blockIdx.x * 8 + (threadIdx.x >> 5);
  if (e >= E) return;
  const int lane = threadIdx.x & 31;
  const int s0 = esrc[e];
  const int j  = esrc[s0];                // double gather
  const int t  = etgt[e];
  const float logit = p_src[j] + p_tgt[t];
  const float lr = logit > 0.f ? logit : 0.2f * logit;
  const float att = __expf(lr);
  const float4 gv = ((const float4*)(src_h + (size_t)j * HCH_))[lane];
  float* np = num + (size_t)t * HCH_ + lane * 4;
  atomicAdd(np + 0, gv.x * att);
  atomicAdd(np + 1, gv.y * att);
  atomicAdd(np + 2, gv.z * att);
  atomicAdd(np + 3, gv.w * att);
  if (lane == 0) atomicAdd(den + t, att);
}

// ---------------------------------------------------------------------------
// out = relu(tgt_h + num / (den + eps))
// ---------------------------------------------------------------------------
__global__ __launch_bounds__(256) void combine_relu(
    const float* __restrict__ tgt_h, const float* __restrict__ num,
    const float* __restrict__ den, float* __restrict__ out, int N)
{
  const size_t i = (size_t)blockIdx.x * 256 + threadIdx.x;
  if (i >= (size_t)N * HCH_) return;
  const int n = (int)(i >> 7);
  const float v = tgt_h[i] + num[i] / (den[n] + 1e-6f);
  out[i] = v > 0.f ? v : 0.f;
}

// ---------------------------------------------------------------------------
extern "C" void kernel_launch(void* const* d_in, const int* in_sizes, int n_in,
                              void* d_out, int out_size, void* d_ws, size_t ws_size,
                              hipStream_t stream)
{
  (void)in_sizes; (void)n_in; (void)out_size; (void)ws_size;

  const float* x_user = (const float*)d_in[0];
  const float* x_spot = (const float*)d_in[1];
  const int*   e_us   = (const int*)d_in[2];
  const int*   e_su   = (const int*)d_in[3];
  const float* Ws_us0 = (const float*)d_in[4];
  const float* Wt_us0 = (const float*)d_in[5];
  const float* a_us0  = (const float*)d_in[6];
  const float* Ws_su0 = (const float*)d_in[7];
  const float* Wt_su0 = (const float*)d_in[8];
  const float* a_su0  = (const float*)d_in[9];
  const float* Ws_us1 = (const float*)d_in[10];
  const float* Wt_us1 = (const float*)d_in[11];
  const float* a_us1  = (const float*)d_in[12];
  const float* Ws_su1 = (const float*)d_in[13];
  const float* Wt_su1 = (const float*)d_in[14];
  const float* a_su1  = (const float*)d_in[15];
  const float* W_ou   = (const float*)d_in[16];
  const float* b_ou   = (const float*)d_in[17];
  const float* W_os   = (const float*)d_in[18];
  const float* b_os   = (const float*)d_in[19];

  const size_t NUH = (size_t)NU_ * HCH_, NSH = (size_t)NS_ * HCH_;

  float* ws    = (float*)d_ws;
  float* hA    = ws;              // user-src transform (u->s edges)
  float* hB    = hA + NUH;        // spot-tgt transform (u->s)
  float* hC    = hB + NSH;        // spot-src transform (s->u)
  float* hD    = hC + NSH;        // user-tgt transform (s->u)
  float* num_s = hD + NUH;        // [NS,H]
  float* den_s = num_s + NSH;     // [NS]
  float* num_u = den_s + NS_;     // [NU,H]
  float* den_u = num_u + NUH;     // [NU]
  float* ps_us = den_u + NU_;
  float* pt_us = ps_us + NU_;
  float* ps_su = pt_us + NS_;
  float* pt_su = ps_su + NS_;
  float* xu    = pt_su + NU_;     // layer-0 output (user)
  float* xs    = xu + NUH;        // layer-0 output (spot)

  float* o_xu = (float*)d_out;                // final xu [NU,128]
  float* o_xs = o_xu + NUH;                   // final xs [NS,128]
  float* o_u  = o_xs + NSH;                   // out_user [NU,64]
  float* o_s  = o_u + (size_t)NU_ * 64;       // out_spot [NS,64]

  const dim3 blk(256);
  const int GB  = (NU_ + 127) / 128;          // NU_ == NS_
  const int GN  = (NU_ + 7) / 8;
  const int GE  = (EDG_ + 7) / 8;
  const int GC  = (int)((NUH + 255) / 256);
  const size_t zbytes = (NSH + NS_ + NUH + NU_) * sizeof(float);

  // ================= layer 0 (K = 64) =================
  gemm_bf16_wmma<64, 128><<<GB, blk, 0, stream>>>(x_user, Ws_us0, nullptr, hA, NU_);
  gemm_bf16_wmma<64, 128><<<GB, blk, 0, stream>>>(x_spot, Wt_us0, nullptr, hB, NS_);
  gemm_bf16_wmma<64, 128><<<GB, blk, 0, stream>>>(x_spot, Ws_su0, nullptr, hC, NS_);
  gemm_bf16_wmma<64, 128><<<GB, blk, 0, stream>>>(x_user, Wt_su0, nullptr, hD, NU_);
  nodedot<<<GN, blk, 0, stream>>>(hA, a_us0,       ps_us, NU_);
  nodedot<<<GN, blk, 0, stream>>>(hB, a_us0 + 128, pt_us, NS_);
  nodedot<<<GN, blk, 0, stream>>>(hC, a_su0,       ps_su, NS_);
  nodedot<<<GN, blk, 0, stream>>>(hD, a_su0 + 128, pt_su, NU_);
  hipMemsetAsync(num_s, 0, zbytes, stream);
  edge_scatter<<<GE, blk, 0, stream>>>(e_us, e_us + EDG_, hA, ps_us, pt_us, num_s, den_s, EDG_);
  edge_scatter<<<GE, blk, 0, stream>>>(e_su, e_su + EDG_, hC, ps_su, pt_su, num_u, den_u, EDG_);
  combine_relu<<<GC, blk, 0, stream>>>(hB, num_s, den_s, xs, NS_);
  combine_relu<<<GC, blk, 0, stream>>>(hD, num_u, den_u, xu, NU_);

  // ================= layer 1 (K = 128) =================
  gemm_bf16_wmma<128, 128><<<GB, blk, 0, stream>>>(xu, Ws_us1, nullptr, hA, NU_);
  gemm_bf16_wmma<128, 128><<<GB, blk, 0, stream>>>(xs, Wt_us1, nullptr, hB, NS_);
  gemm_bf16_wmma<128, 128><<<GB, blk, 0, stream>>>(xs, Ws_su1, nullptr, hC, NS_);
  gemm_bf16_wmma<128, 128><<<GB, blk, 0, stream>>>(xu, Wt_su1, nullptr, hD, NU_);
  nodedot<<<GN, blk, 0, stream>>>(hA, a_us1,       ps_us, NU_);
  nodedot<<<GN, blk, 0, stream>>>(hB, a_us1 + 128, pt_us, NS_);
  nodedot<<<GN, blk, 0, stream>>>(hC, a_su1,       ps_su, NS_);
  nodedot<<<GN, blk, 0, stream>>>(hD, a_su1 + 128, pt_su, NU_);
  hipMemsetAsync(num_s, 0, zbytes, stream);
  edge_scatter<<<GE, blk, 0, stream>>>(e_us, e_us + EDG_, hA, ps_us, pt_us, num_s, den_s, EDG_);
  edge_scatter<<<GE, blk, 0, stream>>>(e_su, e_su + EDG_, hC, ps_su, pt_su, num_u, den_u, EDG_);
  combine_relu<<<GC, blk, 0, stream>>>(hB, num_s, den_s, o_xs, NS_);
  combine_relu<<<GC, blk, 0, stream>>>(hD, num_u, den_u, o_xu, NU_);

  // ================= final per-type linear (128 -> 64, + bias) =================
  gemm_bf16_wmma<128, 64><<<GB, blk, 0, stream>>>(o_xu, W_ou, b_ou, o_u, NU_);
  gemm_bf16_wmma<128, 64><<<GB, blk, 0, stream>>>(o_xs, W_os, b_os, o_s, NS_);
}
